// AttentionPooling_V_15960098472037
// MI455X (gfx1250) — matrix operand order repaired and checked by reading, not compiled, not run on hardware
//
#include <hip/hip_runtime.h>
#include <math.h>

typedef __attribute__((ext_vector_type(16))) _Float16 v16h;
typedef __attribute__((ext_vector_type(8)))  _Float16 v8h;
typedef __attribute__((ext_vector_type(8)))  float    v8f;

#define B_   8
#define N_   8192
#define F_   128
#define H_   64
#define NS_  1024
#define K_   27

// d_out layout (flat float elements, outputs in return order)
#define XPOOL_OFF 0
#define XVN_OFF   (B_*NS_*F_)                  // x_pool:  8*1024*128
#define XS_OFF    (XVN_OFF + B_*NS_*3)         // x_v_next: 8*1024*3
#define PIDX_OFF  (XS_OFF + B_*N_)             // x_s:      8*8192
#define UIDX_OFF  (PIDX_OFF + B_*NS_*K_)       // pooling_idx: 8*1024*27

// ---------------------------------------------------------------------------
// Kernel 1: scores = sigmoid(tanh(x@w1 + b1)@v + bv) via WMMA f16->f32.
// Block = 128 threads = 4 waves; each wave owns a 16-row tile (block: 64 rows).
// w1 is staged into LDS pre-swizzled into per-lane fragment order so each
// B fragment is two contiguous ds_load_b128 (no scalar u16 gathers).
// ---------------------------------------------------------------------------
__global__ __launch_bounds__(128)
void mlp_score_kernel(const float* __restrict__ x,
                      const float* __restrict__ w1,
                      const float* __restrict__ b1,
                      const float* __restrict__ v,
                      const float* __restrict__ bv,
                      float* __restrict__ xs)
{
    // sWs[ks][n][hi][j] : halves j=0..15 of the B fragment for K-step ks,
    // column n, half-wave hi.  B layout (32x16): lane covers column n=lane%16,
// half j -> K = ks*32 + (j<8 ? hi*8 + j : 16 + hi*8 + (j-8)).
    __shared__ __align__(32) _Float16 sWs[4 * H_ * 2 * 16];   // 16 KB
    __shared__ __align__(32) _Float16 sA[4][16 * F_];         // 16 KB (per-wave 16x128)

    const int tid  = threadIdx.x;
    const int wave = tid >> 5;
    const int lane = tid & 31;
    const int hi   = lane >> 4;    // half-wave select
    const int ln   = lane & 15;
    const long rowBase = (long)blockIdx.x * 64 + wave * 16;

    // stage w1 swizzled into fragment order
    for (int e = tid; e < 4 * H_ * 2 * 16; e += 128) {
        const int j  = e & 15;
        const int h2 = (e >> 4) & 1;
        const int n  = (e >> 5) & 63;
        const int ks = e >> 11;
        const int k  = ks * 32 + (j < 8 ? h2 * 8 + j : 16 + h2 * 8 + (j - 8));
        sWs[e] = (_Float16)w1[k * H_ + n];
    }
    // stage 16x128 x tile as f16 (row-major; A fragments read it with b128s)
    {
        const float* xp = x + rowBase * F_;
        for (int e = lane; e < 16 * F_; e += 32) sA[wave][e] = (_Float16)xp[e];
    }
    __syncthreads();

    v8f acc[4];
    #pragma unroll
    for (int t = 0; t < 4; ++t)
        acc[t] = (v8f){0.f,0.f,0.f,0.f,0.f,0.f,0.f,0.f};

    // K = 128 in 4 steps of 32; N = 64 in 4 tiles of 16.
    #pragma unroll
    for (int ks = 0; ks < 4; ++ks) {
        const int kb = ks * 32;
        // A fragment (ISA 7.12.2, 16-bit A 16x32):
        //   halves 0-7  -> K = kb + hi*8 + j      (contiguous, 16B aligned)
        //   halves 8-15 -> K = kb + 16 + hi*8 + j (contiguous, 16B aligned)
        const v8h alo = *(const v8h*)&sA[wave][ln * F_ + kb + hi * 8];
        const v8h ahi = *(const v8h*)&sA[wave][ln * F_ + kb + 16 + hi * 8];
        const v16h a = __builtin_shufflevector(alo, ahi,
                          0,1,2,3,4,5,6,7,8,9,10,11,12,13,14,15);

        #pragma unroll
        for (int t = 0; t < 4; ++t) {
            const int n = t * 16 + ln;
            const v16h bm = *(const v16h*)&sWs[((ks * H_ + n) * 2 + hi) * 16];
            acc[t] = __builtin_amdgcn_wmma_f32_16x16x32_f16(
                         false, a, false, bm, (short)0, acc[t], false, false);
        }
    }

    // h = tanh(acc + b1); partial[r] = sum over this lane's N-columns of h*v
    float partial[8];
    #pragma unroll
    for (int r = 0; r < 8; ++r) partial[r] = 0.f;
    #pragma unroll
    for (int t = 0; t < 4; ++t) {
        const int n    = t * 16 + ln;
        const float bb = b1[n];
        const float vv = v[n];
        #pragma unroll
        for (int r = 0; r < 8; ++r) {
            float h = tanhf(acc[t][r] + bb);
            partial[r] += h * vv;
        }
    }
    // reduce across the 16 lanes of each half-wave (row M = r + 8*hi)
    #pragma unroll
    for (int r = 0; r < 8; ++r) {
        #pragma unroll
        for (int off = 1; off < 16; off <<= 1)
            partial[r] += __shfl_xor(partial[r], off, 16);
    }
    if (ln == 0) {
        const float bvv = bv[0];
        #pragma unroll
        for (int r = 0; r < 8; ++r) {
            const int m = r + 8 * hi;
            const float y = partial[r] + bvv;
            xs[rowBase + m] = 1.f / (1.f + expf(-y));
        }
    }
}

// ---------------------------------------------------------------------------
// Kernel 2: stable top-1024 by score (descending). rank(i) = #{j: s_j>s_i or
// (s_j==s_i and j<i)} reproduces jnp.argsort(-s) exactly. One block per batch.
// ---------------------------------------------------------------------------
__global__ __launch_bounds__(1024)
void topk_select_kernel(const float* __restrict__ xs,
                        const float* __restrict__ xv,
                        float* __restrict__ xvn)
{
    __shared__ float ss[N_];   // 32 KB
    const int b   = blockIdx.x;
    const int tid = threadIdx.x;
    const float* s = xs + (long)b * N_;
    for (int e = tid; e < N_; e += 1024) ss[e] = s[e];
    __syncthreads();

    for (int i = tid; i < N_; i += 1024) {
        const float si = ss[i];
        int rank = 0;
        for (int j = 0; j < N_; ++j) {
            const float sj = ss[j];
            rank += (int)((sj > si) || (sj == si && j < i));
        }
        if (rank < NS_) {
            const float* p = xv  + ((long)b * N_  + i) * 3;
            float*       q = xvn + ((long)b * NS_ + rank) * 3;
            q[0] = p[0]; q[1] = p[1]; q[2] = p[2];
        }
    }
}

// ---------------------------------------------------------------------------
// Kernels 3/4: brute-force KNN with register-resident top-27 (unrolled
// compare-swap chain; strict '<' keeps lower index first on ties, matching
// top_k). Candidate points staged through LDS in chunks.
// ---------------------------------------------------------------------------
#define CH 2048

__global__ __launch_bounds__(256)
void knn_pool_kernel(const float* __restrict__ xv,    // points (B,N,3)
                     const float* __restrict__ xvn,   // queries (B,Ns,3)
                     float* __restrict__ pidx)        // (B,Ns,K) as float
{
    __shared__ float px[CH], py[CH], pz[CH];          // 24 KB
    const int tid = threadIdx.x;
    const int gq  = blockIdx.x * 256 + tid;           // block-uniform batch
    const int b   = gq / NS_;

    const float qx = xvn[(long)gq * 3 + 0];
    const float qy = xvn[(long)gq * 3 + 1];
    const float qz = xvn[(long)gq * 3 + 2];

    float bd[K_]; int bi[K_];
    #pragma unroll
    for (int k = 0; k < K_; ++k) { bd[k] = 3.0e38f; bi[k] = 0; }

    const float* pts = xv + (long)b * N_ * 3;
    for (int c = 0; c < N_; c += CH) {
        __syncthreads();
        for (int e = tid; e < CH; e += 256) {
            px[e] = pts[(c + e) * 3 + 0];
            py[e] = pts[(c + e) * 3 + 1];
            pz[e] = pts[(c + e) * 3 + 2];
        }
        if (c + CH < N_) __builtin_prefetch(&pts[(c + CH) * 3], 0, 1);
        __syncthreads();
        for (int j = 0; j < CH; ++j) {
            const float dx = qx - px[j], dy = qy - py[j], dz = qz - pz[j];
            const float d2 = dx * dx + dy * dy + dz * dz;
            if (d2 < bd[K_ - 1]) {
                bd[K_ - 1] = d2; bi[K_ - 1] = c + j;
                #pragma unroll
                for (int k = K_ - 1; k > 0; --k)
                    if (bd[k] < bd[k - 1]) {
                        float td = bd[k]; bd[k] = bd[k - 1]; bd[k - 1] = td;
                        int   ti = bi[k]; bi[k] = bi[k - 1]; bi[k - 1] = ti;
                    }
            }
        }
    }
    float* out = pidx + (long)gq * K_;
    #pragma unroll
    for (int k = 0; k < K_; ++k) out[k] = (float)bi[k];
}

__global__ __launch_bounds__(256)
void knn_unpool_kernel(const float* __restrict__ xv,   // queries (B,N,3)
                       const float* __restrict__ xvn,  // centers (B,Ns,3)
                       float* __restrict__ uidx)       // (B,N,K) as float
{
    __shared__ float cx[NS_], cy[NS_], cz[NS_];        // 12 KB
    const int tid = threadIdx.x;
    const int gi  = blockIdx.x * 256 + tid;            // block-uniform batch
    const int b   = gi / N_;

    const float* ctr = xvn + (long)b * NS_ * 3;
    for (int e = tid; e < NS_; e += 256) {
        cx[e] = ctr[e * 3 + 0];
        cy[e] = ctr[e * 3 + 1];
        cz[e] = ctr[e * 3 + 2];
    }
    __syncthreads();

    const float qx = xv[(long)gi * 3 + 0];
    const float qy = xv[(long)gi * 3 + 1];
    const float qz = xv[(long)gi * 3 + 2];

    float bd[K_]; int bi[K_];
    #pragma unroll
    for (int k = 0; k < K_; ++k) { bd[k] = 3.0e38f; bi[k] = 0; }

    for (int j = 0; j < NS_; ++j) {
        const float dx = qx - cx[j], dy = qy - cy[j], dz = qz - cz[j];
        const float d2 = dx * dx + dy * dy + dz * dz;
        if (d2 < bd[K_ - 1]) {
            bd[K_ - 1] = d2; bi[K_ - 1] = j;
            #pragma unroll
            for (int k = K_ - 1; k > 0; --k)
                if (bd[k] < bd[k - 1]) {
                    float td = bd[k]; bd[k] = bd[k - 1]; bd[k - 1] = td;
                    int   ti = bi[k]; bi[k] = bi[k - 1]; bi[k - 1] = ti;
                }
        }
    }
    float* out = uidx + (long)gi * K_;
    #pragma unroll
    for (int k = 0; k < K_; ++k) out[k] = (float)bi[k];
}

// ---------------------------------------------------------------------------
// Kernel 5: x_pool[b,q,f] = sum_k (s_k / sum_k s_k) * x[b, idx_k, f]
// One block per pooled point, 128 threads = one per feature.
// ---------------------------------------------------------------------------
__global__ __launch_bounds__(128)
void pool_aggregate_kernel(const float* __restrict__ x,
                           const float* __restrict__ xs,
                           const float* __restrict__ pidx,
                           float* __restrict__ xpool)
{
    __shared__ int   sid[K_];
    __shared__ float ssc[K_];
    const int gq = blockIdx.x;            // b*NS + q
    const int b  = gq / NS_;
    const int f  = threadIdx.x;

    if (f < K_) {
        const int id = (int)pidx[(long)gq * K_ + f];
        sid[f] = id;
        ssc[f] = xs[(long)b * N_ + id];
    }
    __syncthreads();

    float ssum = 0.f;
    #pragma unroll
    for (int k = 0; k < K_; ++k) ssum = ssum + ssc[k];

    float acc = 0.f;
    for (int k = 0; k < K_; ++k)
        acc += ssc[k] * x[((long)b * N_ + sid[k]) * F_ + f];

    xpool[(long)gq * F_ + f] = acc / ssum;
}

// ---------------------------------------------------------------------------
extern "C" void kernel_launch(void* const* d_in, const int* in_sizes, int n_in,
                              void* d_out, int out_size, void* d_ws, size_t ws_size,
                              hipStream_t stream)
{
    (void)in_sizes; (void)n_in; (void)out_size; (void)d_ws; (void)ws_size;

    const float* x  = (const float*)d_in[0];
    const float* xv = (const float*)d_in[1];
    const float* w1 = (const float*)d_in[2];
    const float* b1 = (const float*)d_in[3];
    const float* v  = (const float*)d_in[4];
    const float* bv = (const float*)d_in[5];

    float* out   = (float*)d_out;
    float* xpool = out + XPOOL_OFF;
    float* xvn   = out + XVN_OFF;
    float* xs    = out + XS_OFF;
    float* pidx  = out + PIDX_OFF;
    float* uidx  = out + UIDX_OFF;

    // 1) scores (WMMA MLP)
    mlp_score_kernel<<<(B_ * N_) / 64, 128, 0, stream>>>(x, w1, b1, v, bv, xs);
    // 2) stable top-1024 -> x_v_next
    topk_select_kernel<<<B_, 1024, 0, stream>>>(xs, xv, xvn);
    // 3) pooling_idx: 27-NN of pooled centers among all points
    knn_pool_kernel<<<(B_ * NS_) / 256, 256, 0, stream>>>(xv, xvn, pidx);
    // 4) unpooling_idx: 27-NN of all points among pooled centers
    knn_unpool_kernel<<<(B_ * N_) / 256, 256, 0, stream>>>(xv, xvn, uidx);
    // 5) attention-weighted pooling
    pool_aggregate_kernel<<<B_ * NS_, 128, 0, stream>>>(x, xs, pidx, xpool);
}
